// ODE_RNN_38173669327221
// MI455X (gfx1250) — matrix operand order repaired and checked by reading, not compiled, not run on hardware
//
#include <hip/hip_runtime.h>
#include <hip/hip_bf16.h>

// Problem constants (match reference)
#define L1   512
#define LL2  1024
#define LL3  1536
#define BB   1024
#define TT   64
#define NTHR 512   // 16 wave32 per WG -> fills one WGP (4 waves/SIMD)

// Shared-memory arena (bytes): max(mlp: 32K A + 64K T + 4K red, gru: 16K A + 96K gi)
#define SMEM_BYTES 114688

typedef __bf16 bf16t;
typedef __attribute__((ext_vector_type(16))) __bf16 v16bf;
typedef __attribute__((ext_vector_type(8)))  __bf16 v8bf;
typedef __attribute__((ext_vector_type(8)))  float  v8f;

// Explicit global address space so loads lower to global_load_* (not flat_*,
// which would tie up DScnt + the LDS pipe alongside our ds_load A-fragments).
#define AS1 __attribute__((address_space(1)))
typedef const AS1 float* gcf;
typedef       AS1 float* gf;
typedef const AS1 bf16t* gcb;

static __device__ inline v8f zero_v8f() {
  v8f z = {0.f,0.f,0.f,0.f,0.f,0.f,0.f,0.f};
  return z;
}

static __device__ inline v8f wmma_bf16(v16bf a, v16bf b, v8f c) {
  // (neg_a, A, neg_b, B, c_mod, C, reuse_a, reuse_b)
  return __builtin_amdgcn_wmma_f32_16x16x32_bf16(false, a, false, b, (short)0, c, false, false);
}

// A fragment 16x32 bf16 from LDS, row-major [16][ldk].
// Layout: lanes 0-15 -> M=lane, K base 0; lanes 16-31 -> M=lane-16, K base 16.
// VGPR elems 0..7 -> K = kk+base+0..7 ; elems 8..15 -> K = kk+base+8..15.
static __device__ inline v16bf load_a_lds(const bf16t* __restrict__ sA, int ldk, int kk, int lane) {
  const int m    = lane & 15;
  const int base = (lane < 16) ? 0 : 16;
  v8bf lo = *(const v8bf*)(sA + m * ldk + kk + base);
  v8bf hi = *(const v8bf*)(sA + m * ldk + kk + 8 + base);
  v16bf a;
#pragma unroll
  for (int i = 0; i < 8; ++i) { a[i] = lo[i]; a[8 + i] = hi[i]; }
  return a;
}

// B fragment 32x16 bf16 for D = A @ W^T: B[k][n] = W[nBase+n][k], W row-major [out][in].
// lanes 0-15 -> N=lane, K base 0; lanes 16-31 -> N=lane-16, K base 16; 16 contiguous bf16 per lane.
static __device__ inline v16bf load_b_wt(gcb W, int ldk, int nBase, int kk, int lane) {
  const int n  = lane & 15;
  const int kb = (lane < 16) ? 0 : 16;
  return *(const AS1 v16bf*)(W + (size_t)(nBase + n) * ldk + kk + kb);
}

// K-loop with double-buffered B tiles (prefetch next-K while WMMAs issue).
template<int NT>
static __device__ inline void gemm16_db(v8f* acc, const bf16t* sA, int lda, int K,
                                        gcb W, int ldw, const int* colBase, int lane) {
  v16bf bc[NT];
#pragma unroll
  for (int n = 0; n < NT; ++n) bc[n] = load_b_wt(W, ldw, colBase[n], 0, lane);
  for (int kk = 0; kk < K; kk += 32) {
    v16bf a = load_a_lds(sA, lda, kk, lane);
    const int kn = (kk + 32 < K) ? (kk + 32) : kk;
    v16bf bn[NT];
#pragma unroll
    for (int n = 0; n < NT; ++n) bn[n] = load_b_wt(W, ldw, colBase[n], kn, lane);
#pragma unroll
    for (int n = 0; n < NT; ++n) acc[n] = wmma_bf16(a, bc[n], acc[n]);
#pragma unroll
    for (int n = 0; n < NT; ++n) bc[n] = bn[n];
  }
}

// ---------------- fused f(): leaky(LN(leaky(LN(y@W0^T+b0))@W1^T+b1)) ----------------
// Input combine yEff = y0 + c1*ka + c2*kb + c3*kc folded into the A load.
// If foldSum: write outY = y0 + (ka + 3*(kb+kc) + f(yEff)) * h/8 (RK4 update), else write outK = f(yEff).
static __device__ void mlp_phase(char* smem, int row0, int tid, int w, int lane,
    gcf y0, gcf ka, gcf kb, gcf kc,
    float c1, float c2, float c3,
    gcb W0b, gcf b0, gcf g0, gcf be0,
    gcb W1b, gcf b1, gcf g1, gcf be1,
    gf outK, gf outY, int foldSum)
{
  bf16t* sA  = (bf16t*)smem;                       // 16 x 1024 bf16 (32 KB)
  float* sT  = (float*)(smem + 32768);             // 16 x 1024 f32  (64 KB)
  float* sR1 = (float*)(smem + 32768 + 65536);     // 16 x 32 partials
  float* sR2 = sR1 + 512;
  const int colLane = lane & 15;
  const int rowOff  = (lane >= 16) ? 8 : 0;

  for (int idx = tid; idx < 16 * L1; idx += NTHR) {
    int r = idx >> 9, c = idx & (L1 - 1);
    int g = (row0 + r) * L1 + c;
    float v = y0[g] + c1 * ka[g] + c2 * kb[g] + c3 * kc[g];
    sA[r * LL2 + c] = (bf16t)v;
  }
  __syncthreads();

  // GEMM1: (16x512) @ W0^T -> (16x1024). Wave w owns cols [w*64, w*64+64): 4 tiles.
  {
    v8f acc[4]; int cb[4];
#pragma unroll
    for (int n = 0; n < 4; ++n) { acc[n] = zero_v8f(); cb[n] = (w << 6) + (n << 4); }
    gemm16_db<4>(acc, sA, LL2, L1, W0b, L1, cb, lane);
#pragma unroll
    for (int n = 0; n < 4; ++n) {
      int col = cb[n] + colLane;
      float bias = b0[col];
#pragma unroll
      for (int i = 0; i < 8; ++i) sT[(i + rowOff) * LL2 + col] = acc[n][i] + bias;
    }
  }
  __syncthreads();

  // LN1 + leaky -> sA (bf16, 16x1024). 32 threads per row.
  {
    const int r = tid >> 5, li = tid & 31;
    float s = 0.f, s2 = 0.f;
    for (int c = li; c < LL2; c += 32) { float v = sT[r * LL2 + c]; s += v; s2 += v * v; }
    sR1[r * 32 + li] = s; sR2[r * 32 + li] = s2;
    __syncthreads();
    float ts = 0.f, ts2 = 0.f;
#pragma unroll
    for (int j = 0; j < 32; ++j) { ts += sR1[r * 32 + j]; ts2 += sR2[r * 32 + j]; }
    float mean = ts * (1.f / LL2);
    float var  = ts2 * (1.f / LL2) - mean * mean;
    float inv  = rsqrtf(var + 1e-5f);
    for (int c = li; c < LL2; c += 32) {
      float v = (sT[r * LL2 + c] - mean) * inv * g0[c] + be0[c];
      v = (v >= 0.f) ? v : 0.01f * v;
      sA[r * LL2 + c] = (bf16t)v;
    }
  }
  __syncthreads();

  // GEMM2: (16x1024) @ W1^T -> (16x512). Wave w owns cols [w*32, w*32+32): 2 tiles.
  {
    v8f acc[2]; int cb[2];
#pragma unroll
    for (int n = 0; n < 2; ++n) { acc[n] = zero_v8f(); cb[n] = (w << 5) + (n << 4); }
    gemm16_db<2>(acc, sA, LL2, LL2, W1b, LL2, cb, lane);
#pragma unroll
    for (int n = 0; n < 2; ++n) {
      int col = cb[n] + colLane;
      float bias = b1[col];
#pragma unroll
      for (int i = 0; i < 8; ++i) sT[(i + rowOff) * L1 + col] = acc[n][i] + bias;
    }
  }
  __syncthreads();

  // LN2 + leaky -> k_i, or RK4-fold -> y_next
  {
    const int r = tid >> 5, li = tid & 31;
    float s = 0.f, s2 = 0.f;
    for (int c = li; c < L1; c += 32) { float v = sT[r * L1 + c]; s += v; s2 += v * v; }
    sR1[r * 32 + li] = s; sR2[r * 32 + li] = s2;
    __syncthreads();
    float ts = 0.f, ts2 = 0.f;
#pragma unroll
    for (int j = 0; j < 32; ++j) { ts += sR1[r * 32 + j]; ts2 += sR2[r * 32 + j]; }
    float mean = ts * (1.f / L1);
    float var  = ts2 * (1.f / L1) - mean * mean;
    float inv  = rsqrtf(var + 1e-5f);
    for (int c = li; c < L1; c += 32) {
      float v = (sT[r * L1 + c] - mean) * inv * g1[c] + be1[c];
      v = (v >= 0.f) ? v : 0.01f * v;
      int g = (row0 + r) * L1 + c;
      if (foldSum) outY[g] = y0[g] + (ka[g] + 3.f * (kb[g] + kc[g]) + v) * 0.0625f;
      else         outK[g] = v;
    }
  }
  __syncthreads();  // phase boundary: smem reuse + global k/y visibility inside WG
}

// ---------------- fused GRU step ----------------
static __device__ void gru_phase(char* smem, int row0, int tid, int w, int lane,
    gcf zin, int zinZero,
    gcf xt,      // base = x + tIdx*L1, row stride TT*L1
    gcb Wihb, gcf bih, gcb Whhb, gcf bhh,
    gf zout)
{
  bf16t* sA  = (bf16t*)smem;             // 16 x 512 bf16 (16 KB)
  float* sGi = (float*)(smem + 16384);   // 16 x 1536 f32 (96 KB)
  const int xs = TT * L1;
  const int colLane = lane & 15;
  const int rowOff  = (lane >= 16) ? 8 : 0;

  for (int idx = tid; idx < 16 * L1; idx += NTHR) {
    int r = idx >> 9, c = idx & (L1 - 1);
    float v = zinZero ? 0.f : zin[(row0 + r) * L1 + c];
    sA[r * L1 + c] = (bf16t)v;
  }
  __syncthreads();

  // pass 1: gi = zin @ W_ih^T + b_ih. Wave w owns cols [w*96, w*96+96): 6 tiles.
  {
    v8f acc[6]; int cb[6];
#pragma unroll
    for (int n = 0; n < 6; ++n) { acc[n] = zero_v8f(); cb[n] = w * 96 + (n << 4); }
    gemm16_db<6>(acc, sA, L1, L1, Wihb, L1, cb, lane);
#pragma unroll
    for (int n = 0; n < 6; ++n) {
      int col = cb[n] + colLane;
      float bi = bih[col];
#pragma unroll
      for (int i = 0; i < 8; ++i) sGi[(i + rowOff) * LL3 + col] = acc[n][i] + bi;
    }
  }
  __syncthreads();

  for (int idx = tid; idx < 16 * L1; idx += NTHR) {
    int r = idx >> 9, c = idx & (L1 - 1);
    sA[r * L1 + c] = (bf16t)xt[(row0 + r) * xs + c];
  }
  __syncthreads();

  // pass 2: gh tiles for cols [w*32, w*32+32) in each of the 3 gates, then fuse gates.
  {
    v8f acc[6]; int cb[6];
#pragma unroll
    for (int g = 0; g < 3; ++g)
#pragma unroll
      for (int n = 0; n < 2; ++n) { acc[g * 2 + n] = zero_v8f(); cb[g * 2 + n] = g * L1 + (w << 5) + (n << 4); }
    gemm16_db<6>(acc, sA, L1, L1, Whhb, L1, cb, lane);
#pragma unroll
    for (int n = 0; n < 2; ++n) {
      int col = (w << 5) + (n << 4) + colLane;
      float bhr = bhh[col], bhz = bhh[col + L1], bhn = bhh[col + 2 * L1];
#pragma unroll
      for (int i = 0; i < 8; ++i) {
        int r = i + rowOff, gr = row0 + r;
        float ir  = sGi[r * LL3 + col];
        float iz  = sGi[r * LL3 + col + L1];
        float in_ = sGi[r * LL3 + col + 2 * L1];
        float hr = acc[0 * 2 + n][i] + bhr;
        float hz = acc[1 * 2 + n][i] + bhz;
        float hn = acc[2 * 2 + n][i] + bhn;
        float rg = 1.f / (1.f + __expf(-(ir + hr)));
        float zg = 1.f / (1.f + __expf(-(iz + hz)));
        float ng = tanhf(in_ + rg * hn);
        float hx = xt[gr * xs + col];
        zout[gr * L1 + col] = (1.f - zg) * ng + zg * hx;
      }
    }
  }
  __syncthreads();  // phase boundary
}

// ---------------- head: mu = z@Wm^T+bm ; var = tanh(clip(z@Wv^T+bv)) ----------------
static __device__ void head_phase(char* smem, int row0, int tid, int w, int lane,
    gcf z, gcb Wmb, gcf bm, gcb Wvb, gcf bv, gf out)
{
  bf16t* sA = (bf16t*)smem;
  const int colLane = lane & 15;
  const int rowOff  = (lane >= 16) ? 8 : 0;

  for (int idx = tid; idx < 16 * L1; idx += NTHR) {
    int r = idx >> 9, c = idx & (L1 - 1);
    sA[r * L1 + c] = (bf16t)z[(row0 + r) * L1 + c];
  }
  __syncthreads();

  v8f am[2], av[2]; int cb[2];
#pragma unroll
  for (int n = 0; n < 2; ++n) { am[n] = zero_v8f(); av[n] = zero_v8f(); cb[n] = (w << 5) + (n << 4); }
  gemm16_db<2>(am, sA, L1, L1, Wmb, L1, cb, lane);
  gemm16_db<2>(av, sA, L1, L1, Wvb, L1, cb, lane);
#pragma unroll
  for (int n = 0; n < 2; ++n) {
    int col = cb[n] + colLane;
    float bmc = bm[col], bvc = bv[col];
#pragma unroll
    for (int i = 0; i < 8; ++i) {
      int gr = row0 + i + rowOff;
      out[gr * L1 + col] = am[n][i] + bmc;
      float t = av[n][i] + bvc;
      t = fminf(fmaxf(t, -100.f), 85.f);
      out[BB * L1 + gr * L1 + col] = tanhf(t);
    }
  }
}

// ---------------- persistent scan kernel: one WG owns 16 batch rows for all T ----------------
__global__ __launch_bounds__(NTHR) void k_scan(
    const float* __restrict__ xp,
    const bf16t* __restrict__ Wihp, const float* __restrict__ bihp,
    const bf16t* __restrict__ Whhp, const float* __restrict__ bhhp,
    const bf16t* __restrict__ W0p,  const float* __restrict__ b0p,
    const float* __restrict__ g0p,  const float* __restrict__ be0p,
    const bf16t* __restrict__ W1p,  const float* __restrict__ b1p,
    const float* __restrict__ g1p,  const float* __restrict__ be1p,
    const bf16t* __restrict__ Wmp,  const float* __restrict__ bmp,
    const bf16t* __restrict__ Wvp,  const float* __restrict__ bvp,
    float* __restrict__ zp, float* __restrict__ k1p, float* __restrict__ k2p,
    float* __restrict__ k3p, float* __restrict__ yAp, float* __restrict__ yBp,
    float* __restrict__ outp)
{
  __shared__ alignas(32) char smem[SMEM_BYTES];
  const int tid  = threadIdx.x;
  const int w    = tid >> 5;
  const int lane = tid & 31;
  const int row0 = blockIdx.x << 4;
  const float h  = 0.5f;

  // One explicit generic->global cast per pointer; everything below uses AS1.
  gcf x   = (gcf)xp;
  gcb Wih = (gcb)Wihp; gcf bih = (gcf)bihp;
  gcb Whh = (gcb)Whhp; gcf bhh = (gcf)bhhp;
  gcb W0  = (gcb)W0p;  gcf b0  = (gcf)b0p;  gcf g0  = (gcf)g0p;  gcf be0 = (gcf)be0p;
  gcb W1  = (gcb)W1p;  gcf b1  = (gcf)b1p;  gcf g1  = (gcf)g1p;  gcf be1 = (gcf)be1p;
  gcb Wm  = (gcb)Wmp;  gcf bm  = (gcf)bmp;
  gcb Wv  = (gcb)Wvp;  gcf bv  = (gcf)bvp;
  gf z = (gf)zp; gf k1 = (gf)k1p; gf k2 = (gf)k2p; gf k3 = (gf)k3p;
  gf yA = (gf)yAp; gf yB = (gf)yBp; gf out = (gf)outp;

  // z0 = GRU(0, x[:, T-1, :])
  gru_phase(smem, row0, tid, w, lane, (gcf)nullptr, 1, x + (size_t)(TT - 1) * L1,
            Wih, bih, Whh, bhh, z);

#pragma unroll 1
  for (int t = 0; t < TT; ++t) {
    gcf xt = x + (size_t)(TT - 1 - t) * L1;

    // rk4_step(z, 0.5) -> yA   (k4 eval folds the RK4 update)
    mlp_phase(smem, row0, tid, w, lane, z, z, z, z, 0.f, 0.f, 0.f,
              W0, b0, g0, be0, W1, b1, g1, be1, k1, (gf)nullptr, 0);
    mlp_phase(smem, row0, tid, w, lane, z, k1, z, z, h / 3.f, 0.f, 0.f,
              W0, b0, g0, be0, W1, b1, g1, be1, k2, (gf)nullptr, 0);
    mlp_phase(smem, row0, tid, w, lane, z, k1, k2, z, -h / 3.f, h, 0.f,
              W0, b0, g0, be0, W1, b1, g1, be1, k3, (gf)nullptr, 0);
    mlp_phase(smem, row0, tid, w, lane, z, k1, k2, k3, h, -h, h,
              W0, b0, g0, be0, W1, b1, g1, be1, (gf)nullptr, yA, 1);

    // rk4_step(yA, 0.5) -> yB
    mlp_phase(smem, row0, tid, w, lane, yA, yA, yA, yA, 0.f, 0.f, 0.f,
              W0, b0, g0, be0, W1, b1, g1, be1, k1, (gf)nullptr, 0);
    mlp_phase(smem, row0, tid, w, lane, yA, k1, yA, yA, h / 3.f, 0.f, 0.f,
              W0, b0, g0, be0, W1, b1, g1, be1, k2, (gf)nullptr, 0);
    mlp_phase(smem, row0, tid, w, lane, yA, k1, k2, yA, -h / 3.f, h, 0.f,
              W0, b0, g0, be0, W1, b1, g1, be1, k3, (gf)nullptr, 0);
    mlp_phase(smem, row0, tid, w, lane, yA, k1, k2, k3, h, -h, h,
              W0, b0, g0, be0, W1, b1, g1, be1, (gf)nullptr, yB, 1);

    // z = GRU(yB, x_t)
    gru_phase(smem, row0, tid, w, lane, yB, 0, xt, Wih, bih, Whh, bhh, z);
  }

  head_phase(smem, row0, tid, w, lane, z, Wm, bm, Wv, bv, out);
}

// ---------------- weight conversion ----------------
__global__ void k_cvt_bf16(const float* __restrict__ s, bf16t* __restrict__ d, int n) {
  int i = blockIdx.x * blockDim.x + threadIdx.x;
  if (i < n) ((AS1 bf16t*)d)[i] = (bf16t)((gcf)s)[i];
}

// ---------------- host launch ----------------
extern "C" void kernel_launch(void* const* d_in, const int* in_sizes, int n_in,
                              void* d_out, int out_size, void* d_ws, size_t ws_size,
                              hipStream_t stream)
{
  (void)in_sizes; (void)n_in; (void)out_size; (void)ws_size;
  const float* x   = (const float*)d_in[0];
  const float* Wih = (const float*)d_in[1];
  const float* bih = (const float*)d_in[2];
  const float* Whh = (const float*)d_in[3];
  const float* bhh = (const float*)d_in[4];
  const float* W0  = (const float*)d_in[5];
  const float* b0  = (const float*)d_in[6];
  const float* g0  = (const float*)d_in[7];
  const float* be0 = (const float*)d_in[8];
  const float* W1  = (const float*)d_in[9];
  const float* b1  = (const float*)d_in[10];
  const float* g1  = (const float*)d_in[11];
  const float* be1 = (const float*)d_in[12];
  const float* Wm  = (const float*)d_in[13];
  const float* bm  = (const float*)d_in[14];
  const float* Wv  = (const float*)d_in[15];
  const float* bv  = (const float*)d_in[16];
  float* outp = (float*)d_out;

  char* base = (char*)d_ws;
  size_t off = 0;
  auto take = [&](size_t bytes) -> void* {
    void* p = base + off;
    off += bytes;
    off = (off + 255) & ~(size_t)255;
    return p;
  };
  bf16t* W0b  = (bf16t*)take((size_t)LL2 * L1 * 2);
  bf16t* W1b  = (bf16t*)take((size_t)L1 * LL2 * 2);
  bf16t* Wihb = (bf16t*)take((size_t)LL3 * L1 * 2);
  bf16t* Whhb = (bf16t*)take((size_t)LL3 * L1 * 2);
  bf16t* Wmb  = (bf16t*)take((size_t)L1 * L1 * 2);
  bf16t* Wvb  = (bf16t*)take((size_t)L1 * L1 * 2);
  const size_t NZ = (size_t)BB * L1;
  float* z  = (float*)take(NZ * 4);
  float* k1 = (float*)take(NZ * 4);
  float* k2 = (float*)take(NZ * 4);
  float* k3 = (float*)take(NZ * 4);
  float* yA = (float*)take(NZ * 4);
  float* yB = (float*)take(NZ * 4);

  auto cvt = [&](const float* s, bf16t* d, int n) {
    k_cvt_bf16<<<(n + 255) / 256, 256, 0, stream>>>(s, d, n);
  };
  cvt(W0,  W0b,  LL2 * L1);
  cvt(W1,  W1b,  L1 * LL2);
  cvt(Wih, Wihb, LL3 * L1);
  cvt(Whh, Whhb, LL3 * L1);
  cvt(Wm,  Wmb,  L1 * L1);
  cvt(Wv,  Wvb,  L1 * L1);

  // Persistent scan: 64 WGs x 512 threads (16 waves), each WG owns 16 batch rows end-to-end.
  k_scan<<<BB / 16, NTHR, 0, stream>>>(
      x, Wihb, bih, Whhb, bhh,
      W0b, b0, g0, be0, W1b, b1, g1, be1,
      Wmb, bm, Wvb, bv,
      z, k1, k2, k3, yA, yB, outp);
}
